// DMPNNNet_90134183674523
// MI455X (gfx1250) — matrix operand order repaired and checked by reading, not compiled, not run on hardware
//
#include <hip/hip_runtime.h>
#include <hip/hip_bf16.h>
#include <cstdint>

typedef __attribute__((ext_vector_type(16))) _Float16 v16h;
typedef __attribute__((ext_vector_type(8)))  _Float16 v8h;
typedef __attribute__((ext_vector_type(8)))  float    v8f;
typedef __attribute__((ext_vector_type(4)))  float    v4f;

__device__ __forceinline__ v8f wmma16x16x32(v16h a, v16h b, v8f c) {
  // D = A(16x32 f16) * B(32x16 f16) + C(16x16 f32)
  return __builtin_amdgcn_wmma_f32_16x16x32_f16(false, a, false, b, (short)0, c,
                                                false, false);
}

// ---------------------------------------------------------------------------
// Stage `nbytes` of global data into LDS with CDNA5 async copies
// (GLOBAL_LOAD_ASYNC_TO_LDS_B128, tracked by ASYNCcnt). nbytes must be a
// multiple of 16 and fully covered by the 256-thread block loop.
// LDS byte offset = low 32 bits of the generic pointer (aperture scheme).
// ---------------------------------------------------------------------------
__device__ __forceinline__ void stage_async(const void* gsrc, void* ldst,
                                            int nbytes, int t) {
  const char* g = (const char*)gsrc;
  uint32_t lbase = (uint32_t)(uintptr_t)ldst;
  for (int i = t * 16; i < nbytes; i += 256 * 16) {
    uint64_t ga = (uint64_t)(uintptr_t)(g + i);
    uint32_t la = lbase + (uint32_t)i;
    asm volatile("global_load_async_to_lds_b128 %0, %1, off"
                 :: "v"(la), "v"(ga) : "memory");
  }
  asm volatile("s_wait_asynccnt 0x0" ::: "memory");
}

// ---------------------------------------------------------------------------
// node_h = leaky_relu(node @ Wn + bn)   [N,40]x[40,64]; writes f32 + f16 copy
// ---------------------------------------------------------------------------
__global__ void node_embed_kernel(const float* __restrict__ node,
                                  const float* __restrict__ Wn,
                                  const float* __restrict__ bn,
                                  float* __restrict__ node_h,
                                  _Float16* __restrict__ node_h16,
                                  int N) {
  __shared__ float sW[40 * 64];
  __shared__ float sB[64];
  int t = threadIdx.x;
  for (int i = t; i < 40 * 64; i += 256) sW[i] = Wn[i];
  if (t < 64) sB[t] = bn[t];
  __syncthreads();
  int idx = blockIdx.x * 256 + t;
  if (idx >= N * 64) return;
  int n = idx >> 6, k = idx & 63;
  const float* nr = node + (size_t)n * 40;
  float acc = sB[k];
#pragma unroll
  for (int j = 0; j < 40; ++j) acc = fmaf(nr[j], sW[j * 64 + k], acc);
  acc = acc > 0.f ? acc : 0.01f * acc;  // leaky_relu(0.01)
  node_h[idx] = acc;
  node_h16[idx] = (_Float16)acc;
}

// ---------------------------------------------------------------------------
// edge_h16 = f16(leaky_relu(edge @ We + be))   [E,10]x[10,64]
// ---------------------------------------------------------------------------
__global__ void edge_embed_kernel(const float* __restrict__ edge,
                                  const float* __restrict__ We,
                                  const float* __restrict__ be,
                                  _Float16* __restrict__ edge_h16,
                                  int E) {
  __shared__ float sW[10 * 64];
  __shared__ float sB[64];
  int t = threadIdx.x;
  for (int i = t; i < 10 * 64; i += 256) sW[i] = We[i];
  if (t < 64) sB[t] = be[t];
  __syncthreads();
  int idx = blockIdx.x * 256 + t;
  if (idx >= E * 64) return;
  int e = idx >> 6, k = idx & 63;
  const float* er = edge + (size_t)e * 10;
  float acc = sB[k];
#pragma unroll
  for (int j = 0; j < 10; ++j) acc = fmaf(er[j], sW[j * 64 + k], acc);
  acc = acc > 0.f ? acc : 0.01f * acc;
  edge_h16[idx] = (_Float16)acc;
}

// ---------------------------------------------------------------------------
// h = inputs = relu(concat(node_h[src], edge_h) @ Wh + bh)
// WMMA: M=16 edges/wave, K=128 (4 k-steps), N=64 (4 tiles). Wh^T f16 in LDS.
// ---------------------------------------------------------------------------
__global__ void dmpnn_init_kernel(const _Float16* __restrict__ node_h16,
                                  const _Float16* __restrict__ edge_h16,
                                  const int* __restrict__ src,
                                  const float* __restrict__ Wh,
                                  const float* __restrict__ bh,
                                  float* __restrict__ h,
                                  float* __restrict__ inputs,
                                  int E) {
  __shared__ __align__(128) float    sStage[128 * 64];  // raw Wh, 32 KB
  __shared__ __align__(128) _Float16 sWT[64 * 128];     // [n][k] f16, 16 KB
  __shared__ float sB[64];
  int t = threadIdx.x;
  stage_async(Wh, sStage, 128 * 64 * 4, t);
  if (t < 64) sB[t] = bh[t];
  __syncthreads();
  for (int i = t; i < 64 * 128; i += 256) {
    int n = i >> 7, k = i & 127;
    sWT[i] = (_Float16)sStage[k * 64 + n];  // transpose: WT[n][k] = Wh[k][n]
  }
  __syncthreads();

  int wave = t >> 5, lane = t & 31;
  int hf = lane >> 4, mrow = lane & 15;
  int base = blockIdx.x * 128 + wave * 16;
  int eA = base + mrow;
  int ec = eA < E ? eA : E - 1;
  const _Float16* nrow = node_h16 + (size_t)src[ec] * 64;
  const _Float16* erow = edge_h16 + (size_t)ec * 64;

  v16h a[4];
#pragma unroll
  for (int ks = 0; ks < 4; ++ks) {
    const _Float16* rp = (ks < 2) ? nrow : erow;
    int k0 = ((ks & 1) * 32) + hf * 8;
    v8h lo = *(const v8h*)(rp + k0);
    v8h hi = *(const v8h*)(rp + k0 + 16);
#pragma unroll
    for (int j = 0; j < 8; ++j) { a[ks][j] = lo[j]; a[ks][8 + j] = hi[j]; }
  }

  // ks-outer / nt-inner: consecutive WMMAs hit independent accumulators
  v8f acc[4] = {{}, {}, {}, {}};
#pragma unroll
  for (int ks = 0; ks < 4; ++ks) {
#pragma unroll
    for (int nt = 0; nt < 4; ++nt) {
      const _Float16* bp = &sWT[(nt * 16 + mrow) * 128 + ks * 32 + hf * 8];
      v8h blo = *(const v8h*)bp, bhi = *(const v8h*)(bp + 16);
      v16h bf = __builtin_shufflevector(blo, bhi, 0, 1, 2, 3, 4, 5, 6, 7, 8, 9, 10, 11, 12, 13, 14, 15);
      acc[nt] = wmma16x16x32(a[ks], bf, acc[nt]);
    }
  }

  bool fullTile = (base + 16 <= E);  // wave-uniform
#pragma unroll
  for (int nt = 0; nt < 4; ++nt) {
    int col = nt * 16 + mrow;
    float bb = sB[col];
    size_t rbase = (size_t)(base + hf * 8) * 64 + col;
    float* hp = h + rbase;
    float* ip = inputs + rbase;
    if (fullTile) {
#pragma unroll
      for (int v = 0; v < 8; ++v) {
        float val = acc[nt][v] + bb;
        val = val > 0.f ? val : 0.f;
        hp[v * 64] = val;
        ip[v * 64] = val;
      }
    } else {
#pragma unroll
      for (int v = 0; v < 8; ++v) {
        if (base + hf * 8 + v < E) {
          float val = acc[nt][v] + bb;
          val = val > 0.f ? val : 0.f;
          hp[v * 64] = val;
          ip[v * 64] = val;
        }
      }
    }
  }
}

// ---------------------------------------------------------------------------
// One DMPNN layer (in-place on h):
//   h = relu(inputs + (full[dst] - h) @ Wm_d + bm_d)
// msg -> f16 fused into A-fragment build; Wm_d^T f16 in LDS. Each wave reads
// and writes exactly its own 16 edge rows, so in-place update is race-free.
// ---------------------------------------------------------------------------
__global__ void dmpnn_layer_kernel(float* __restrict__ h,
                                   const float* __restrict__ inputs,
                                   const float* __restrict__ full,
                                   const int* __restrict__ dstIdx,
                                   const float* __restrict__ W,   // [64,64]
                                   const float* __restrict__ bias,
                                   int E) {
  __shared__ __align__(128) float    sStage[64 * 64];  // raw W, 16 KB
  __shared__ __align__(128) _Float16 sWT[64 * 64];     // [n][k] f16, 8 KB
  __shared__ float sB[64];
  int t = threadIdx.x;
  stage_async(W, sStage, 64 * 64 * 4, t);
  if (t < 64) sB[t] = bias[t];
  __syncthreads();
  for (int i = t; i < 64 * 64; i += 256) {
    int n = i >> 6, k = i & 63;
    sWT[i] = (_Float16)sStage[k * 64 + n];
  }
  __syncthreads();

  int wave = t >> 5, lane = t & 31;
  int hf = lane >> 4, mrow = lane & 15;
  int base = blockIdx.x * 128 + wave * 16;
  int eA = base + mrow;
  int ec = eA < E ? eA : E - 1;
  const float* frow = full + (size_t)dstIdx[ec] * 64;  // L2-resident gather
  const float* hrow = h + (size_t)ec * 64;

  v16h a[2];
#pragma unroll
  for (int ks = 0; ks < 2; ++ks) {
    int k0 = ks * 32 + hf * 8;
    v4f f0 = *(const v4f*)(frow + k0);
    v4f f1 = *(const v4f*)(frow + k0 + 4);
    v4f f2 = *(const v4f*)(frow + k0 + 16);
    v4f f3 = *(const v4f*)(frow + k0 + 20);
    v4f g0 = *(const v4f*)(hrow + k0);
    v4f g1 = *(const v4f*)(hrow + k0 + 4);
    v4f g2 = *(const v4f*)(hrow + k0 + 16);
    v4f g3 = *(const v4f*)(hrow + k0 + 20);
#pragma unroll
    for (int j = 0; j < 4; ++j) {
      a[ks][j]      = (_Float16)(f0[j] - g0[j]);
      a[ks][4 + j]  = (_Float16)(f1[j] - g1[j]);
      a[ks][8 + j]  = (_Float16)(f2[j] - g2[j]);
      a[ks][12 + j] = (_Float16)(f3[j] - g3[j]);
    }
  }

  v8f acc[4] = {{}, {}, {}, {}};
#pragma unroll
  for (int ks = 0; ks < 2; ++ks) {
#pragma unroll
    for (int nt = 0; nt < 4; ++nt) {
      const _Float16* bp = &sWT[(nt * 16 + mrow) * 64 + ks * 32 + hf * 8];
      v8h blo = *(const v8h*)bp, bhi = *(const v8h*)(bp + 16);
      v16h bf = __builtin_shufflevector(blo, bhi, 0, 1, 2, 3, 4, 5, 6, 7, 8, 9, 10, 11, 12, 13, 14, 15);
      acc[nt] = wmma16x16x32(a[ks], bf, acc[nt]);
    }
  }

  bool fullTile = (base + 16 <= E);
#pragma unroll
  for (int nt = 0; nt < 4; ++nt) {
    int col = nt * 16 + mrow;
    float bb = sB[col];
    size_t rbase = (size_t)(base + hf * 8) * 64 + col;
    float* hp = h + rbase;
    const float* ip = inputs + rbase;
    if (fullTile) {
#pragma unroll
      for (int v = 0; v < 8; ++v) {
        float val = ip[v * 64] + acc[nt][v] + bb;
        hp[v * 64] = val > 0.f ? val : 0.f;
      }
    } else {
#pragma unroll
      for (int v = 0; v < 8; ++v) {
        if (base + hf * 8 + v < E) {
          float val = ip[v * 64] + acc[nt][v] + bb;
          hp[v * 64] = val > 0.f ? val : 0.f;
        }
      }
    }
  }
}

// ---------------------------------------------------------------------------
// node_hidden = relu(concat(node_h, incoming) @ Wa + ba)   K=128 WMMA
// ---------------------------------------------------------------------------
__global__ void node_hidden_kernel(const _Float16* __restrict__ node_h16,
                                   const float* __restrict__ incoming,
                                   const float* __restrict__ Wa,
                                   const float* __restrict__ ba,
                                   float* __restrict__ node_hidden,
                                   int N) {
  __shared__ __align__(128) float    sStage[128 * 64];
  __shared__ __align__(128) _Float16 sWT[64 * 128];
  __shared__ float sB[64];
  int t = threadIdx.x;
  stage_async(Wa, sStage, 128 * 64 * 4, t);
  if (t < 64) sB[t] = ba[t];
  __syncthreads();
  for (int i = t; i < 64 * 128; i += 256) {
    int n = i >> 7, k = i & 127;
    sWT[i] = (_Float16)sStage[k * 64 + n];
  }
  __syncthreads();

  int wave = t >> 5, lane = t & 31;
  int hf = lane >> 4, mrow = lane & 15;
  int base = blockIdx.x * 128 + wave * 16;
  int nd = base + mrow;
  int nc = nd < N ? nd : N - 1;
  const _Float16* nrow = node_h16 + (size_t)nc * 64;
  const float* irow = incoming + (size_t)nc * 64;

  v16h a[4];
#pragma unroll
  for (int ks = 0; ks < 4; ++ks) {
    int k0 = ((ks & 1) * 32) + hf * 8;
    if (ks < 2) {
      v8h lo = *(const v8h*)(nrow + k0);
      v8h hi = *(const v8h*)(nrow + k0 + 16);
#pragma unroll
      for (int j = 0; j < 8; ++j) { a[ks][j] = lo[j]; a[ks][8 + j] = hi[j]; }
    } else {
      v4f f0 = *(const v4f*)(irow + k0);
      v4f f1 = *(const v4f*)(irow + k0 + 4);
      v4f f2 = *(const v4f*)(irow + k0 + 16);
      v4f f3 = *(const v4f*)(irow + k0 + 20);
#pragma unroll
      for (int j = 0; j < 4; ++j) {
        a[ks][j]      = (_Float16)f0[j];
        a[ks][4 + j]  = (_Float16)f1[j];
        a[ks][8 + j]  = (_Float16)f2[j];
        a[ks][12 + j] = (_Float16)f3[j];
      }
    }
  }

  v8f acc[4] = {{}, {}, {}, {}};
#pragma unroll
  for (int ks = 0; ks < 4; ++ks) {
#pragma unroll
    for (int nt = 0; nt < 4; ++nt) {
      const _Float16* bp = &sWT[(nt * 16 + mrow) * 128 + ks * 32 + hf * 8];
      v8h blo = *(const v8h*)bp, bhi = *(const v8h*)(bp + 16);
      v16h bf = __builtin_shufflevector(blo, bhi, 0, 1, 2, 3, 4, 5, 6, 7, 8, 9, 10, 11, 12, 13, 14, 15);
      acc[nt] = wmma16x16x32(a[ks], bf, acc[nt]);
    }
  }

  bool fullTile = (base + 16 <= N);
#pragma unroll
  for (int nt = 0; nt < 4; ++nt) {
    int col = nt * 16 + mrow;
    float bb = sB[col];
    float* op = node_hidden + (size_t)(base + hf * 8) * 64 + col;
    if (fullTile) {
#pragma unroll
      for (int v = 0; v < 8; ++v) {
        float val = acc[nt][v] + bb;
        op[v * 64] = val > 0.f ? val : 0.f;
      }
    } else {
#pragma unroll
      for (int v = 0; v < 8; ++v) {
        if (base + hf * 8 + v < N) {
          float val = acc[nt][v] + bb;
          op[v * 64] = val > 0.f ? val : 0.f;
        }
      }
    }
  }
}

// ---------------------------------------------------------------------------
// helpers: zero / segment scatter-adds (global_atomic_add_f32)
// ---------------------------------------------------------------------------
__global__ void zero_kernel(float* __restrict__ p, int n) {
  int i = blockIdx.x * 256 + threadIdx.x;
  if (i < n) p[i] = 0.f;
}

__global__ void scatter_add_kernel(const float* __restrict__ v,
                                   const int* __restrict__ idx,
                                   float* __restrict__ out, int E) {
  int i = blockIdx.x * 256 + threadIdx.x;
  if (i >= E * 64) return;
  int e = i >> 6, k = i & 63;
  atomicAdd(out + (size_t)idx[e] * 64 + k, v[i]);
}

__global__ void pool_kernel(const float* __restrict__ node_hidden,
                            const int* __restrict__ gids,
                            float* __restrict__ gf, int N) {
  int i = blockIdx.x * 256 + threadIdx.x;
  if (i >= N * 64) return;
  int n = i >> 6, k = i & 63;
  atomicAdd(gf + (size_t)gids[n] * 64 + k, node_hidden[i]);
}

// ---------------------------------------------------------------------------
// out[g] = relu(gf[g] @ Wp1 + bp1) @ Wp2 + bp2     (tiny: 2000x(64x32+32))
// ---------------------------------------------------------------------------
__global__ void readout_kernel(const float* __restrict__ gf,
                               const float* __restrict__ Wp1,
                               const float* __restrict__ bp1,
                               const float* __restrict__ Wp2,
                               const float* __restrict__ bp2,
                               float* __restrict__ out, int G) {
  __shared__ float sW1[64 * 32];
  __shared__ float sB1[32];
  __shared__ float sW2[32];
  __shared__ float sB2;
  int t = threadIdx.x;  // 64 threads
  for (int i = t; i < 64 * 32; i += 64) sW1[i] = Wp1[i];
  if (t < 32) { sB1[t] = bp1[t]; sW2[t] = Wp2[t]; }
  if (t == 0) sB2 = bp2[0];
  __syncthreads();
  int g = blockIdx.x * 64 + t;
  if (g >= G) return;
  const float* row = gf + (size_t)g * 64;
  float acc = sB2;
#pragma unroll 4
  for (int m = 0; m < 32; ++m) {
    float hs = sB1[m];
#pragma unroll
    for (int j = 0; j < 64; ++j) hs = fmaf(row[j], sW1[j * 32 + m], hs);
    hs = hs > 0.f ? hs : 0.f;
    acc = fmaf(hs, sW2[m], acc);
  }
  out[g] = acc;
}

// ---------------------------------------------------------------------------
extern "C" void kernel_launch(void* const* d_in, const int* in_sizes, int n_in,
                              void* d_out, int out_size, void* d_ws, size_t ws_size,
                              hipStream_t stream) {
  const float* node = (const float*)d_in[0];
  const float* edge = (const float*)d_in[1];
  const float* Wn  = (const float*)d_in[2];
  const float* bn  = (const float*)d_in[3];
  const float* We  = (const float*)d_in[4];
  const float* be  = (const float*)d_in[5];
  const float* Wh  = (const float*)d_in[6];
  const float* bh  = (const float*)d_in[7];
  const float* Wm  = (const float*)d_in[8];   // [4,64,64]
  const float* bm  = (const float*)d_in[9];   // [4,64]
  const float* Wa  = (const float*)d_in[10];
  const float* ba  = (const float*)d_in[11];
  const float* Wp1 = (const float*)d_in[12];
  const float* bp1 = (const float*)d_in[13];
  const float* Wp2 = (const float*)d_in[14];
  const float* bp2 = (const float*)d_in[15];
  const int* src  = (const int*)d_in[16];
  const int* dstI = (const int*)d_in[17];
  const int* gids = (const int*)d_in[18];
  (void)n_in; (void)ws_size;

  const int N = in_sizes[0] / 40;   // 50000
  const int E = in_sizes[1] / 10;   // 800000
  const int G = out_size;           // 2000 (out is [G,1])

  // Workspace carve-out (all buffers fully rewritten each call).
  char* ws = (char*)d_ws;
  size_t off = 0;
  auto carve = [&](size_t bytes) -> void* {
    void* p = (void*)(ws + off);
    off += (bytes + 255) & ~(size_t)255;
    return p;
  };
  float*    node_h      = (float*)   carve((size_t)N * 64 * sizeof(float));
  _Float16* node_h16    = (_Float16*)carve((size_t)N * 64 * sizeof(_Float16));
  _Float16* edge_h16    = (_Float16*)carve((size_t)E * 64 * sizeof(_Float16));
  float*    hbuf        = (float*)   carve((size_t)E * 64 * sizeof(float));
  float*    inputs      = (float*)   carve((size_t)E * 64 * sizeof(float));
  float*    full        = (float*)   carve((size_t)N * 64 * sizeof(float));
  float*    node_hidden = (float*)   carve((size_t)N * 64 * sizeof(float));
  float*    gf          = (float*)   carve((size_t)G * 64 * sizeof(float));
  (void)node_h;

  const int NV = N * 64;  // 3.2M
  const int EV = E * 64;  // 51.2M

  node_embed_kernel<<<(NV + 255) / 256, 256, 0, stream>>>(node, Wn, bn, node_h,
                                                          node_h16, N);
  edge_embed_kernel<<<(EV + 255) / 256, 256, 0, stream>>>(edge, We, be,
                                                          edge_h16, E);
  dmpnn_init_kernel<<<(E + 127) / 128, 256, 0, stream>>>(node_h16, edge_h16, src,
                                                         Wh, bh, hbuf, inputs, E);
  for (int d = 0; d < 4; ++d) {
    zero_kernel<<<(NV + 255) / 256, 256, 0, stream>>>(full, NV);
    scatter_add_kernel<<<(EV + 255) / 256, 256, 0, stream>>>(hbuf, dstI, full, E);
    dmpnn_layer_kernel<<<(E + 127) / 128, 256, 0, stream>>>(
        hbuf, inputs, full, dstI, Wm + (size_t)d * 64 * 64, bm + (size_t)d * 64, E);
  }
  // readout: incoming edge-sum per node, then node MLP
  zero_kernel<<<(NV + 255) / 256, 256, 0, stream>>>(full, NV);
  scatter_add_kernel<<<(EV + 255) / 256, 256, 0, stream>>>(hbuf, dstI, full, E);
  node_hidden_kernel<<<(N + 127) / 128, 256, 0, stream>>>(node_h16, full, Wa, ba,
                                                          node_hidden, N);
  zero_kernel<<<(G * 64 + 255) / 256, 256, 0, stream>>>(gf, G * 64);
  pool_kernel<<<(NV + 255) / 256, 256, 0, stream>>>(node_hidden, gids, gf, N);
  readout_kernel<<<(G + 63) / 64, 64, 0, stream>>>(gf, Wp1, bp1, Wp2, bp2,
                                                   (float*)d_out, G);
}
